// MockLLMBlock_9380208574867
// MI455X (gfx1250) — matrix operand order repaired and checked by reading, not compiled, not run on hardware
//
#include <hip/hip_runtime.h>
#include <hip/hip_bf16.h>

// ---------------------------------------------------------------------------
// MI455X (gfx1250) transformer block: bf16 WMMA GEMMs + flash attention.
// wave32, WMMA 16x16x32 bf16 -> f32 accum. Weights pre-transposed so every
// LDS tile stage is contiguous 16B traffic, moved by the gfx1250 async
// global->LDS path (GLOBAL_LOAD_ASYNC_TO_LDS_B128, ASYNCcnt).
// ---------------------------------------------------------------------------

typedef __attribute__((ext_vector_type(16))) __bf16 v16bf;
typedef __attribute__((ext_vector_type(8)))  float  v8f;

#define HIDDEN 2048
#define DFF    8192
#define BATCH  2
#define TSEQ   2048
#define TOKENS (BATCH * TSEQ)
#define NHEADS 16
#define HD     128

#define WMMA_BF16(a, b, c) \
  __builtin_amdgcn_wmma_f32_16x16x32_bf16(false, (a), false, (b), (short)0, (c), false, false)

// Low 32 bits of a generic pointer into __shared__ == wave-relative LDS offset
// (flat->LDS mapping truncates to addr[31:0], ISA 10.2).
__device__ __forceinline__ unsigned lds_off(const void* p) {
  return (unsigned)(unsigned long long)p;
}

// Async DMA: 16 bytes global -> LDS per lane (ASYNCcnt-tracked, no VGPR data).
__device__ __forceinline__ void async_copy_b128(unsigned lds, const void* gptr) {
  asm volatile("global_load_async_to_lds_b128 %0, %1, off"
               :: "v"(lds), "v"(gptr) : "memory");
}
__device__ __forceinline__ void wait_async0() {
  asm volatile("s_wait_asynccnt 0x0" ::: "memory");
}

// A fragment (16x32, M x K), bf16, from row-major LDS tile (row stride ld elems).
__device__ __forceinline__ v16bf load_frag_a(const __bf16* base, int ld, int ko) {
  int lane = threadIdx.x & 31;
  const __bf16* p = base + (lane & 15) * ld + ko + ((lane >> 4) << 3);
  v16bf f;
  ((uint4*)&f)[0] = *(const uint4*)p;
  ((uint4*)&f)[1] = *(const uint4*)(p + 16);
  return f;
}

// B fragment (32x16, K x N), bf16, from TRANSPOSED LDS tile Bt[n][k] (row stride ld).
__device__ __forceinline__ v16bf load_frag_b(const __bf16* baseT, int ld, int ko) {
  int lane = threadIdx.x & 31;
  const __bf16* p = baseT + (lane & 15) * ld + ko + ((lane >> 4) << 3);
  v16bf f;
  ((uint4*)&f)[0] = *(const uint4*)p;
  ((uint4*)&f)[1] = *(const uint4*)(p + 16);
  return f;
}

// ---------------------------------------------------------------------------
// Tiled transpose + cast to bf16: in[R][C] (f32 or bf16) -> out[C][R] (bf16).
// ---------------------------------------------------------------------------
template <typename TI>
__global__ __launch_bounds__(256)
void transpose_cast_kernel(const TI* __restrict__ in, __bf16* __restrict__ out,
                           int R, int C) {
  __shared__ __bf16 tile[64][65];
  int r0 = blockIdx.y * 64, c0 = blockIdx.x * 64;
  int t = threadIdx.x;
  {
    int rr = t >> 2, cc = (t & 3) * 16;
    const TI* src = in + (long)(r0 + rr) * C + c0 + cc;
#pragma unroll
    for (int i = 0; i < 16; ++i) tile[rr][cc + i] = (__bf16)(float)src[i];
  }
  __syncthreads();
  {
    int oc = t >> 2, orr = (t & 3) * 16;
    __bf16* dst = out + (long)(c0 + oc) * R + r0 + orr;
#pragma unroll
    for (int i = 0; i < 16; ++i) dst[i] = tile[orr + i][oc];
  }
}

// ---------------------------------------------------------------------------
// LayerNorm over HIDDEN=2048, one block (8 waves) per row, output bf16
// ---------------------------------------------------------------------------
__global__ __launch_bounds__(256)
void ln_cast_kernel(const float* __restrict__ x, const float* __restrict__ w,
                    const float* __restrict__ bias, __bf16* __restrict__ out) {
  const int H = HIDDEN;
  long row = blockIdx.x;
  const float* xr = x + row * H;
  int t = threadIdx.x, wid = t >> 5, lane = t & 31;
  float vals[8];
  float s = 0.f;
#pragma unroll
  for (int i = 0; i < 8; ++i) { vals[i] = xr[t + i * 256]; s += vals[i]; }
#pragma unroll
  for (int m = 16; m >= 1; m >>= 1) s += __shfl_xor(s, m, 32);
  __shared__ float red[8];
  if (lane == 0) red[wid] = s;
  __syncthreads();
  float tot = 0.f;
#pragma unroll
  for (int i = 0; i < 8; ++i) tot += red[i];
  float mu = tot * (1.0f / H);
  float vs = 0.f;
#pragma unroll
  for (int i = 0; i < 8; ++i) { float d = vals[i] - mu; vs += d * d; }
#pragma unroll
  for (int m = 16; m >= 1; m >>= 1) vs += __shfl_xor(vs, m, 32);
  __syncthreads();
  if (lane == 0) red[wid] = vs;
  __syncthreads();
  float vtot = 0.f;
#pragma unroll
  for (int i = 0; i < 8; ++i) vtot += red[i];
  float inv = rsqrtf(vtot * (1.0f / H) + 1e-5f);
#pragma unroll
  for (int i = 0; i < 8; ++i) {
    int c = t + i * 256;
    out[row * H + c] = (__bf16)((vals[i] - mu) * inv * w[c] + bias[c]);
  }
}

// ---------------------------------------------------------------------------
// bf16 WMMA GEMM: C[M,N] = A[M,K] @ Bt^T, with Bt = B transposed [N][K].
// 128x128 block tile, KTILE=64, 8 waves each own 64x32 (4x2 WMMA tiles).
// Async global->LDS DMA with ping-pong buffers: ONE barrier per K-tile;
// tile i+1 DMA overlaps tile i WMMAs.
// ---------------------------------------------------------------------------
enum { EPI_BF16 = 0, EPI_RESID_F32 = 1, EPI_BIAS_SILU_BF16 = 2, EPI_BIAS_RESID_F32 = 3 };

template <int EPI>
__global__ __launch_bounds__(256)
void gemm_bf16_kernel(const __bf16* __restrict__ A, const __bf16* __restrict__ Bt,
                      float* __restrict__ Cf, __bf16* __restrict__ Cb,
                      const float* __restrict__ bias, const float* __restrict__ resid,
                      int M, int N, int K) {
  __shared__ __align__(16) __bf16 At[2][128 * 64];   // [m][k] ping-pong
  __shared__ __align__(16) __bf16 Btp[2][128 * 64];  // [n][k] ping-pong
  int t = threadIdx.x;
  int w = t >> 5, lane = t & 31;
  int wm = w >> 2, wn = w & 3;           // 2 waves along M, 4 along N
  int m0 = blockIdx.y * 128, n0 = blockIdx.x * 128;

  int r = t >> 1, hf = (t & 1) * 32;     // staging coords: 64B per thread per tile
  const __bf16* aSrc = A  + (long)(m0 + r) * K + hf;
  const __bf16* bSrc = Bt + (long)(n0 + r) * K + hf;
  unsigned laBase[2], lbBase[2];
#pragma unroll
  for (int p = 0; p < 2; ++p) {
    laBase[p] = lds_off(&At[p][r * 64 + hf]);
    lbBase[p] = lds_off(&Btp[p][r * 64 + hf]);
  }

  v8f acc[4][2];
#pragma unroll
  for (int i = 0; i < 4; ++i)
#pragma unroll
    for (int j = 0; j < 2; ++j) acc[i][j] = (v8f){0,0,0,0,0,0,0,0};

  // prologue: DMA tile 0 into buffer 0
#pragma unroll
  for (int i = 0; i < 4; ++i) {
    async_copy_b128(laBase[0] + i * 16, aSrc + i * 8);
    async_copy_b128(lbBase[0] + i * 16, bSrc + i * 8);
  }

  int nkt = K >> 6;
  for (int kt = 0; kt < nkt; ++kt) {
    int p = kt & 1;
    wait_async0();        // my DMA for tile kt has landed in LDS
    __syncthreads();      // everyone's landed; other buffer is free
    if (kt + 1 < nkt) {   // DMA tile kt+1 into the free buffer, overlap compute
      int k1 = (kt + 1) << 6;
#pragma unroll
      for (int i = 0; i < 4; ++i) {
        async_copy_b128(laBase[1 - p] + i * 16, aSrc + k1 + i * 8);
        async_copy_b128(lbBase[1 - p] + i * 16, bSrc + k1 + i * 8);
      }
    }
    // compute: 2 k-steps of 32, 8 WMMAs each
#pragma unroll
    for (int kk = 0; kk < 64; kk += 32) {
      v16bf af[4], bfr[2];
#pragma unroll
      for (int i = 0; i < 4; ++i) af[i] = load_frag_a(&At[p][(wm * 64 + i * 16) * 64], 64, kk);
#pragma unroll
      for (int j = 0; j < 2; ++j) bfr[j] = load_frag_b(&Btp[p][(wn * 32 + j * 16) * 64], 64, kk);
#pragma unroll
      for (int i = 0; i < 4; ++i)
#pragma unroll
        for (int j = 0; j < 2; ++j) acc[i][j] = WMMA_BF16(af[i], bfr[j], acc[i][j]);
    }
  }

  // epilogue (C layout: VGPR r -> row r (+8 upper half), col = lane&15)
  int ncol = lane & 15, mofs = (lane >> 4) << 3;
#pragma unroll
  for (int i = 0; i < 4; ++i)
#pragma unroll
    for (int j = 0; j < 2; ++j)
#pragma unroll
      for (int rr = 0; rr < 8; ++rr) {
        long m = m0 + wm * 64 + i * 16 + rr + mofs;
        long n = n0 + wn * 32 + j * 16 + ncol;
        float v = acc[i][j][rr];
        if (EPI == EPI_BIAS_SILU_BF16 || EPI == EPI_BIAS_RESID_F32) v += bias[n];
        if (EPI == EPI_BIAS_SILU_BF16) v = v / (1.0f + __expf(-v));
        if (EPI == EPI_RESID_F32 || EPI == EPI_BIAS_RESID_F32) v += resid[m * N + n];
        if (EPI == EPI_BF16 || EPI == EPI_BIAS_SILU_BF16) Cb[m * N + n] = (__bf16)v;
        else Cf[m * N + n] = v;
      }
}

// ---------------------------------------------------------------------------
// Flash attention: one block per (batch, head, 128 query rows).
// 8 waves x 16 query rows. Q frags live in registers across the KV sweep.
// K and pre-transposed V blocks staged by async global->LDS DMA.
// ---------------------------------------------------------------------------
__global__ __launch_bounds__(256)
void attention_kernel(const __bf16* __restrict__ Q, const __bf16* __restrict__ Kmat,
                      const __bf16* __restrict__ Vt, __bf16* __restrict__ O) {
  const int LD = HIDDEN;
  int blk = blockIdx.x;
  int qb = blk & 15;           // T/128 = 16 query blocks
  int h  = (blk >> 4) & 15;
  int b  = blk >> 8;
  long baseRow = (long)b * TSEQ;
  long colOff  = (long)h * HD;

  __shared__ __align__(16) __bf16 bufA[128 * 128];  // Q staging, then V^T [d][kpos]
  __shared__ __align__(16) __bf16 bufB[128 * 128];  // K [kpos][d], then P

  int t = threadIdx.x, w = t >> 5, lane = t & 31;
  int ncol = lane & 15, mofs = (lane >> 4) << 3;
  int sr = t >> 1, shf = (t & 1) * 64;   // staging coords: 128B per thread
  unsigned laA = lds_off(&bufA[sr * 128 + shf]);
  unsigned laB = lds_off(&bufB[sr * 128 + shf]);

  // ---- stage Q block (row-major 128x128) via async DMA ----
  {
    const __bf16* src = Q + (baseRow + qb * 128 + sr) * LD + colOff + shf;
#pragma unroll
    for (int i = 0; i < 8; ++i) async_copy_b128(laA + i * 16, src + i * 8);
    wait_async0();
  }
  __syncthreads();
  v16bf aq[4];
#pragma unroll
  for (int ks = 0; ks < 4; ++ks) aq[ks] = load_frag_a(bufA + (w * 16) * 128, 128, ks * 32);
  __syncthreads();  // bufA free for V^T

  v8f o_acc[8];
#pragma unroll
  for (int i = 0; i < 8; ++i) o_acc[i] = (v8f){0,0,0,0,0,0,0,0};
  float rm[8], rs[8];
#pragma unroll
  for (int rr = 0; rr < 8; ++rr) { rm[rr] = -1e30f; rs[rr] = 0.f; }
  const float scale = 0.08838834764831845f;  // 1/sqrt(128)

  for (int kb = 0; kb <= qb; ++kb) {
    // ---- async DMA: K block [kpos][d] -> bufB, V^T block [d][kpos] -> bufA ----
    {
      const __bf16* ksrc = Kmat + (baseRow + kb * 128 + sr) * LD + colOff + shf;
      const __bf16* vsrc = Vt + (colOff + sr) * (long)TOKENS + baseRow + kb * 128 + shf;
#pragma unroll
      for (int i = 0; i < 8; ++i) {
        async_copy_b128(laB + i * 16, ksrc + i * 8);
        async_copy_b128(laA + i * 16, vsrc + i * 8);
      }
      wait_async0();
    }
    __syncthreads();

    // ---- scores: 8 n-tiles x 4 k-steps = 32 WMMAs ----
    v8f s[8];
#pragma unroll
    for (int nt = 0; nt < 8; ++nt) {
      v8f a = (v8f){0,0,0,0,0,0,0,0};
#pragma unroll
      for (int ks = 0; ks < 4; ++ks) {
        v16bf bk = load_frag_b(bufB + (nt * 16) * 128, 128, ks * 32);
        a = WMMA_BF16(aq[ks], bk, a);
      }
      s[nt] = a;
    }
    // ---- scale + causal mask (diagonal block only) ----
#pragma unroll
    for (int nt = 0; nt < 8; ++nt)
#pragma unroll
      for (int rr = 0; rr < 8; ++rr) {
        float v = s[nt][rr] * scale;
        if (kb == qb) {
          int key  = kb * 128 + nt * 16 + ncol;
          int qrow = qb * 128 + w * 16 + rr + mofs;
          if (key > qrow) v = -1e30f;
        }
        s[nt][rr] = v;
      }
    // ---- online softmax (rows live in 16-lane halves; xor<16 stays in half) ----
#pragma unroll
    for (int rr = 0; rr < 8; ++rr) {
      float mx = s[0][rr];
#pragma unroll
      for (int nt = 1; nt < 8; ++nt) mx = fmaxf(mx, s[nt][rr]);
#pragma unroll
      for (int m = 8; m >= 1; m >>= 1) mx = fmaxf(mx, __shfl_xor(mx, m, 32));
      float nm = fmaxf(rm[rr], mx);
      float alpha = __expf(rm[rr] - nm);
      rm[rr] = nm;
      float ps = 0.f;
#pragma unroll
      for (int nt = 0; nt < 8; ++nt) { float p = __expf(s[nt][rr] - nm); s[nt][rr] = p; ps += p; }
#pragma unroll
      for (int m = 8; m >= 1; m >>= 1) ps += __shfl_xor(ps, m, 32);
      rs[rr] = rs[rr] * alpha + ps;
#pragma unroll
      for (int dt = 0; dt < 8; ++dt) o_acc[dt][rr] *= alpha;
    }
    __syncthreads();  // all waves done reading K in bufB

    // ---- spill P (16x128 per wave, row-major) into its bufB slice ----
    __bf16* Pw = bufB + w * 16 * 128;
#pragma unroll
    for (int nt = 0; nt < 8; ++nt)
#pragma unroll
      for (int rr = 0; rr < 8; ++rr)
        Pw[(rr + mofs) * 128 + nt * 16 + ncol] = (__bf16)s[nt][rr];

    // ---- P @ V : 4 k-steps x 8 d-tiles = 32 WMMAs (wave-private P, in-order LDS) ----
#pragma unroll
    for (int ks = 0; ks < 4; ++ks) {
      v16bf ap = load_frag_a(Pw, 128, ks * 32);
#pragma unroll
      for (int dt = 0; dt < 8; ++dt) {
        v16bf bv = load_frag_b(bufA + (dt * 16) * 128, 128, ks * 32);
        o_acc[dt] = WMMA_BF16(ap, bv, o_acc[dt]);
      }
    }
    __syncthreads();
  }

  // ---- normalize + write out (bf16, [b,t, h*128+d] layout) ----
#pragma unroll
  for (int dt = 0; dt < 8; ++dt)
#pragma unroll
    for (int rr = 0; rr < 8; ++rr) {
      float val = o_acc[dt][rr] / rs[rr];
      long qrow = qb * 128 + w * 16 + rr + mofs;
      long d = dt * 16 + ncol;
      O[(baseRow + qrow) * LD + colOff + d] = (__bf16)val;
    }
}

// ---------------------------------------------------------------------------
// Host orchestration
// ---------------------------------------------------------------------------
extern "C" void kernel_launch(void* const* d_in, const int* in_sizes, int n_in,
                              void* d_out, int out_size, void* d_ws, size_t ws_size,
                              hipStream_t stream) {
  (void)in_sizes; (void)n_in; (void)out_size; (void)ws_size;
  const float* x    = (const float*)d_in[0];
  // d_in[1] = causal mask (ignored; causality handled analytically)
  const float* Wq   = (const float*)d_in[2];
  const float* Wk   = (const float*)d_in[3];
  const float* Wv   = (const float*)d_in[4];
  const float* Wo   = (const float*)d_in[5];
  const float* ln1w = (const float*)d_in[6];
  const float* ln1b = (const float*)d_in[7];
  const float* ln2w = (const float*)d_in[8];
  const float* ln2b = (const float*)d_in[9];
  const float* W1   = (const float*)d_in[10];
  const float* b1   = (const float*)d_in[11];
  const float* W2   = (const float*)d_in[12];
  const float* b2   = (const float*)d_in[13];

  const long SZ_W   = (long)HIDDEN * HIDDEN;
  const long SZ_WFF = (long)HIDDEN * DFF;
  const long SZ_ACT = (long)TOKENS * HIDDEN;
  const long SZ_MID = (long)TOKENS * DFF;

  char* wp = (char*)d_ws;
  size_t off = 0;
  auto take = [&](size_t bytes) {
    char* p = wp + off;
    off += (bytes + 255) & ~(size_t)255;
    return p;
  };
  __bf16* Wqt  = (__bf16*)take(SZ_W * 2);    // [N][K] bf16
  __bf16* Wkt  = (__bf16*)take(SZ_W * 2);
  __bf16* Wvt  = (__bf16*)take(SZ_W * 2);
  __bf16* Wot  = (__bf16*)take(SZ_W * 2);
  __bf16* W1t  = (__bf16*)take(SZ_WFF * 2);  // [DFF][HIDDEN]
  __bf16* W2t  = (__bf16*)take(SZ_WFF * 2);  // [HIDDEN][DFF]
  __bf16* h1   = (__bf16*)take(SZ_ACT * 2);
  __bf16* qb   = (__bf16*)take(SZ_ACT * 2);
  __bf16* kbuf = (__bf16*)take(SZ_ACT * 2);
  __bf16* vbuf = (__bf16*)take(SZ_ACT * 2);
  __bf16* vtb  = (__bf16*)take(SZ_ACT * 2);  // V transposed [HIDDEN][TOKENS]
  __bf16* ao   = (__bf16*)take(SZ_ACT * 2);
  float*  xmid = (float*) take(SZ_ACT * 4);
  __bf16* h2   = (__bf16*)take(SZ_ACT * 2);
  __bf16* mid  = (__bf16*)take(SZ_MID * 2);

  // 1) weight cast + transpose (one pass; 100MB bf16 then resident in 192MB L2)
  dim3 gW (HIDDEN / 64, HIDDEN / 64);
  dim3 gW1(DFF / 64,    HIDDEN / 64);
  dim3 gW2(HIDDEN / 64, DFF / 64);
  transpose_cast_kernel<float><<<gW,  256, 0, stream>>>(Wq, Wqt, HIDDEN, HIDDEN);
  transpose_cast_kernel<float><<<gW,  256, 0, stream>>>(Wk, Wkt, HIDDEN, HIDDEN);
  transpose_cast_kernel<float><<<gW,  256, 0, stream>>>(Wv, Wvt, HIDDEN, HIDDEN);
  transpose_cast_kernel<float><<<gW,  256, 0, stream>>>(Wo, Wot, HIDDEN, HIDDEN);
  transpose_cast_kernel<float><<<gW1, 256, 0, stream>>>(W1, W1t, HIDDEN, DFF);
  transpose_cast_kernel<float><<<gW2, 256, 0, stream>>>(W2, W2t, DFF, HIDDEN);

  // 2) LN1 -> h1 (bf16)
  ln_cast_kernel<<<TOKENS, 256, 0, stream>>>(x, ln1w, ln1b, h1);

  // 3) Q/K/V projections (M=4096, N=2048, K=2048)
  dim3 gProj(HIDDEN / 128, TOKENS / 128);
  gemm_bf16_kernel<EPI_BF16><<<gProj, 256, 0, stream>>>(h1, Wqt, nullptr, qb,   nullptr, nullptr, TOKENS, HIDDEN, HIDDEN);
  gemm_bf16_kernel<EPI_BF16><<<gProj, 256, 0, stream>>>(h1, Wkt, nullptr, kbuf, nullptr, nullptr, TOKENS, HIDDEN, HIDDEN);
  gemm_bf16_kernel<EPI_BF16><<<gProj, 256, 0, stream>>>(h1, Wvt, nullptr, vbuf, nullptr, nullptr, TOKENS, HIDDEN, HIDDEN);

  // 4) transpose V once (bf16->bf16) so attention stages V^T contiguously
  dim3 gVt(HIDDEN / 64, TOKENS / 64);
  transpose_cast_kernel<__bf16><<<gVt, 256, 0, stream>>>(vbuf, vtb, TOKENS, HIDDEN);

  // 5) flash attention: B*H*(T/128) = 512 blocks
  attention_kernel<<<BATCH * NHEADS * (TSEQ / 128), 256, 0, stream>>>(qb, kbuf, vtb, ao);

  // 6) output projection + residual -> xmid (f32)
  gemm_bf16_kernel<EPI_RESID_F32><<<gProj, 256, 0, stream>>>(ao, Wot, xmid, nullptr, nullptr, x, TOKENS, HIDDEN, HIDDEN);

  // 7) LN2 -> h2 (bf16)
  ln_cast_kernel<<<TOKENS, 256, 0, stream>>>(xmid, ln2w, ln2b, h2);

  // 8) MLP up + SiLU -> mid (bf16)  (M=4096, N=8192, K=2048)
  dim3 gUp(DFF / 128, TOKENS / 128);
  gemm_bf16_kernel<EPI_BIAS_SILU_BF16><<<gUp, 256, 0, stream>>>(h2, W1t, nullptr, mid, b1, nullptr, TOKENS, DFF, HIDDEN);

  // 9) MLP down + bias + residual -> d_out (f32)  (M=4096, N=2048, K=8192)
  gemm_bf16_kernel<EPI_BIAS_RESID_F32><<<gProj, 256, 0, stream>>>(mid, W2t, (float*)d_out, nullptr, b2, xmid, TOKENS, HIDDEN, DFF);
}
